// HgsMultiHeadAttention_46961172414620
// MI455X (gfx1250) — matrix operand order repaired and checked by reading, not compiled, not run on hardware
//
#include <hip/hip_runtime.h>
#include <hip/hip_bf16.h>

typedef __attribute__((ext_vector_type(16))) _Float16 v16h;
typedef __attribute__((ext_vector_type(4)))  _Float16 v4h;
typedef __attribute__((ext_vector_type(8)))  float    v8f;

#define B_  4
#define S_  2048
#define D_  256
#define H_  8
#define T_  4
#define DK_ 32
#define NTOK (B_*S_)

// WMMA wave32 fragment index maps (CDNA5 ISA 7.12.2):
// A 16x32 f16: lane holds row=lane&15; elem i -> K = (i<8 ? i : i+8) + 8*(lane>>4)
__device__ __forceinline__ int kmapA(int i, int hi) { return (i < 8 ? i : i + 8) + 8 * hi; }
// B 32x16 f16: lane holds col=lane&15; elem i -> K = (i<8 ? i : i+8) + 8*(lane>>4)
__device__ __forceinline__ int kmapB(int i, int hi) { return (i < 8 ? i : i + 8) + 8 * hi; }
// C/D f32 16x16: elem j -> M = j + 8*(lane>>4), N = lane&15

// ---------------------------------------------------------------------------
// Prep A: f32 -> f16 conversion for the weight tensors, vectorized x4.
// ---------------------------------------------------------------------------
__global__ __launch_bounds__(256)
void cvt_f16_kernel(const float* __restrict__ src, _Float16* __restrict__ dst, int n4)
{
    const int i = blockIdx.x * 256 + threadIdx.x;
    if (i < n4) {
        const float4 v = ((const float4*)src)[i];
        v4h o;
        o.x = (_Float16)v.x; o.y = (_Float16)v.y;
        o.z = (_Float16)v.z; o.w = (_Float16)v.w;
        ((v4h*)dst)[i] = o;
    }
}

// ---------------------------------------------------------------------------
// Prep B: build 4 type-masked f16 copies of x:
//   xmsk[tt][token][d] = (xtype[token]==tt) ? x[token][d] : 0
// This pre-bakes the per-type routing so the QKV kernel's A fragments are
// plain b128 loads with zero per-element select VALU in the WMMA loop.
// ---------------------------------------------------------------------------
__global__ __launch_bounds__(256)
void xmask_f16_kernel(const float* __restrict__ x, const int* __restrict__ xtype,
                      _Float16* __restrict__ xmsk, int n4)   // n4 = NTOK*D_/4
{
    const int i = blockIdx.x * 256 + threadIdx.x;
    if (i >= n4) return;
    const float4 v = ((const float4*)x)[i];
    const int token = i / (D_ / 4);
    const int t = xtype[token];
    v4h o;
    o.x = (_Float16)v.x; o.y = (_Float16)v.y;
    o.z = (_Float16)v.z; o.w = (_Float16)v.w;
    v4h z = {};
    #pragma unroll
    for (int tt = 0; tt < T_; ++tt)
        ((v4h*)xmsk)[(size_t)tt * n4 + i] = (tt == t) ? o : z;
}

// ---------------------------------------------------------------------------
// Kernel 1: per-type QKV projection, all-f16 operands, no in-loop masking.
// 8 waves / block; each wave computes a 16-token x 16-output tile for q,k,v.
// Stores Q,K as f16 [b,h,s,dk] and V transposed as f16 [b,h,dk,s].
// ---------------------------------------------------------------------------
__global__ __launch_bounds__(256)
void hgs_qkv_kernel(const _Float16* __restrict__ xmsk, const int* __restrict__ xtype,
                    const _Float16* __restrict__ Wqh, const float* __restrict__ bq,
                    const _Float16* __restrict__ Wkh, const float* __restrict__ bk,
                    const _Float16* __restrict__ Wvh, const float* __restrict__ bv,
                    _Float16* __restrict__ qws, _Float16* __restrict__ kws,
                    _Float16* __restrict__ vtws)
{
    const int colTile = blockIdx.x;                      // 0..15
    const int wave = threadIdx.x >> 5;
    const int rowTile = blockIdx.y * 8 + wave;           // 0..511
    const int lane = threadIdx.x & 31;
    const int row  = lane & 15;
    const int hi   = lane >> 4;
    const int token = rowTile * 16 + row;
    const int colBase = colTile * 16;

    const _Float16* Wmat[3] = {Wqh, Wkh, Wvh};
    v8f zero = {};
    v8f acc[3] = {zero, zero, zero};

    for (int kstep = 0; kstep < D_ / 32; ++kstep) {
        const int kBase = kstep * 32;
        #pragma unroll
        for (int tt = 0; tt < T_; ++tt) {
            // A rows pre-masked by type: contiguous b128 loads, no VALU select
            const _Float16* xm = xmsk + (size_t)tt * NTOK * D_ + (size_t)token * D_ + kBase;
            v16h a;
            #pragma unroll
            for (int i = 0; i < 16; ++i) a[i] = xm[kmapA(i, hi)];
            #pragma unroll
            for (int p = 0; p < 3; ++p) {
                // B[k][n] = W[tt][o = colBase+n][d = kBase+k]  (y = W x, torch layout)
                const _Float16* Wp = Wmat[p] + ((size_t)tt * D_ + (colBase + row)) * D_ + kBase;
                v16h bfrag;
                #pragma unroll
                for (int i = 0; i < 16; ++i) bfrag[i] = Wp[kmapB(i, hi)];
                acc[p] = __builtin_amdgcn_wmma_f32_16x16x32_f16(
                    false, a, false, bfrag, (short)0, acc[p], false, false);
            }
        }
    }

    // epilogue: add per-row-type bias, scatter to Q/K (row-major) and V^T (f16)
    #pragma unroll
    for (int j = 0; j < 8; ++j) {
        const int M = j + 8 * hi;
        const int tokM = rowTile * 16 + M;
        const int tM = xtype[tokM];
        const int o = colBase + row;           // 0..255
        const int b = tokM / S_;
        const int s = tokM % S_;
        const int h = o >> 5;
        const int dk = o & 31;
        const float qv = acc[0][j] + bq[tM * D_ + o];
        const float kv = acc[1][j] + bk[tM * D_ + o];
        const float vv = acc[2][j] + bv[tM * D_ + o];
        const size_t qkIdx = ((size_t)(b * H_ + h) * S_ + s) * DK_ + dk;
        qws[qkIdx] = (_Float16)qv;
        kws[qkIdx] = (_Float16)kv;
        vtws[((size_t)(b * H_ + h) * DK_ + dk) * S_ + s] = (_Float16)vv;
    }
}

// ---------------------------------------------------------------------------
// Kernel 2: attention for one (b, h, 16-query row tile).
// phase1: raw QK^T scores -> LDS strip (pure WMMA pipeline)
// phase2: fused mask(int4)+scale+softmax, float4 lanes, single b128 p_attn write
// phase3: ctx = P @ V with ds_add_f32 reduction.
// ---------------------------------------------------------------------------
#define SMEM_FLOATS (16 * S_ + 16 * DK_ + 256 + 16)

__global__ __launch_bounds__(256)
void hgs_attn_kernel(const _Float16* __restrict__ qws, const _Float16* __restrict__ kws,
                     const _Float16* __restrict__ vtws, const int* __restrict__ mask,
                     float* __restrict__ out, float* __restrict__ pattn)
{
    extern __shared__ float smem[];
    float* sc      = smem;                 // [16][S_] score strip
    float* ctxbuf  = sc + 16 * S_;         // [16][32] ctx accumulator
    float* red     = ctxbuf + 16 * DK_;    // [16][16] reduction scratch
    float* rowstat = red + 256;            // [16] row max / row sum

    const int rowTile = blockIdx.x;
    const int h = blockIdx.y;
    const int b = blockIdx.z;
    const int rowBase = rowTile * 16;
    const int tid  = threadIdx.x;
    const int wave = tid >> 5;
    const int lane = tid & 31;
    const int arow = lane & 15;
    const int hi   = lane >> 4;

    const _Float16* Qh = qws  + (size_t)(b * H_ + h) * S_ * DK_;
    const _Float16* Kh = kws  + (size_t)(b * H_ + h) * S_ * DK_;
    const _Float16* Vh = vtws + (size_t)(b * H_ + h) * DK_ * S_;   // [32][S_]
    const int* maskB = mask + (size_t)b * S_ * S_;

    // ---- phase 1: raw scores = Q K^T ; DK=32 -> one WMMA per 16x16 tile ----
    v16h aq;
    #pragma unroll
    for (int i = 0; i < 16; ++i) aq[i] = Qh[(rowBase + arow) * DK_ + kmapA(i, hi)];

    for (int ct = wave; ct < S_ / 16; ct += 8) {
        const int colBase = ct * 16;
        if (ct + 8 < S_ / 16)   // warm L2 for the mask bytes phase 2 will stream
            __builtin_prefetch(&maskB[(size_t)(rowBase + arow) * S_ + (ct + 8) * 16], 0, 1);
        v16h bk;
        #pragma unroll
        for (int i = 0; i < 16; ++i)
            bk[i] = Kh[(colBase + arow) * DK_ + kmapB(i, hi)];  // B[k][n] = K[n][k]
        v8f sv = {};
        sv = __builtin_amdgcn_wmma_f32_16x16x32_f16(false, aq, false, bk,
                                                    (short)0, sv, false, false);
        #pragma unroll
        for (int j = 0; j < 8; ++j)
            sc[(j + 8 * hi) * S_ + colBase + arow] = sv[j];
    }
    __syncthreads();

    // ---- phase 2: mask+scale fused with softmax; 16 threads/row, float4/int4 ----
    const int r = tid >> 4;
    const int g = tid & 15;
    float4* scr = reinterpret_cast<float4*>(sc + r * S_);            // 512 chunks
    const int4* mrow = reinterpret_cast<const int4*>(maskB + (size_t)(rowBase + r) * S_);
    const float rscale = 0.17677669529663687f;  // 1/sqrt(32)

    float lmax = -3.4e38f;
    for (int c4 = g; c4 < S_ / 4; c4 += 16) {
        float4 v = scr[c4];
        const int4 m = mrow[c4];
        v.x = (m.x != 0) ? v.x * rscale : -1e9f;
        v.y = (m.y != 0) ? v.y * rscale : -1e9f;
        v.z = (m.z != 0) ? v.z * rscale : -1e9f;
        v.w = (m.w != 0) ? v.w * rscale : -1e9f;
        scr[c4] = v;
        lmax = fmaxf(lmax, fmaxf(fmaxf(v.x, v.y), fmaxf(v.z, v.w)));
    }
    red[r * 16 + g] = lmax;
    __syncthreads();
    if (g == 0) {
        float m = red[r * 16];
        #pragma unroll
        for (int t2 = 1; t2 < 16; ++t2) m = fmaxf(m, red[r * 16 + t2]);
        rowstat[r] = m;
    }
    __syncthreads();
    const float rmax = rowstat[r];
    float lsum = 0.f;
    for (int c4 = g; c4 < S_ / 4; c4 += 16) {
        float4 v = scr[c4];
        v.x = __expf(v.x - rmax);
        v.y = __expf(v.y - rmax);
        v.z = __expf(v.z - rmax);
        v.w = __expf(v.w - rmax);
        scr[c4] = v;
        lsum += (v.x + v.y) + (v.z + v.w);
    }
    red[r * 16 + g] = lsum;
    ctxbuf[tid] = 0.f;            // zero ctx accumulator while we're here
    ctxbuf[tid + 256] = 0.f;
    __syncthreads();
    if (g == 0) {
        float s2 = 0.f;
        #pragma unroll
        for (int t2 = 0; t2 < 16; ++t2) s2 += red[r * 16 + t2];
        rowstat[r] = s2;
    }
    __syncthreads();
    const float invsum = 1.0f / rowstat[r];
    float* pOut = pattn + ((size_t)(b * H_ + h) * S_ + rowBase) * S_;
    float4* pr = reinterpret_cast<float4*>(pOut + (size_t)r * S_);
    for (int c4 = g; c4 < S_ / 4; c4 += 16) {   // normalize in place + b128 p_attn write
        float4 v = scr[c4];
        v.x *= invsum; v.y *= invsum; v.z *= invsum; v.w *= invsum;
        scr[c4] = v;
        pr[c4] = v;
    }
    __syncthreads();

    // ---- phase 3: ctx = P @ V ; 8 waves split the 64 K-steps ----
    v8f acc0 = {};
    v8f acc1 = {};
    for (int kt = wave; kt < S_ / 32; kt += 8) {
        const int kBase = kt * 32;
        v16h ap, bv0, bv1;
        #pragma unroll
        for (int i = 0; i < 16; ++i) {
            ap[i]  = (_Float16)sc[arow * S_ + kBase + kmapA(i, hi)];
            bv0[i] = Vh[(size_t)arow * S_ + kBase + kmapB(i, hi)];          // d = 0..15
            bv1[i] = Vh[(size_t)(arow + 16) * S_ + kBase + kmapB(i, hi)];   // d = 16..31
        }
        acc0 = __builtin_amdgcn_wmma_f32_16x16x32_f16(false, ap, false, bv0,
                                                      (short)0, acc0, false, false);
        acc1 = __builtin_amdgcn_wmma_f32_16x16x32_f16(false, ap, false, bv1,
                                                      (short)0, acc1, false, false);
    }
    #pragma unroll
    for (int j = 0; j < 8; ++j) {        // cross-wave reduce via LDS float atomics
        const int M = j + 8 * hi;
        atomicAdd(&ctxbuf[M * DK_ + arow], acc0[j]);
        atomicAdd(&ctxbuf[M * DK_ + 16 + arow], acc1[j]);
    }
    __syncthreads();
    for (int idx = tid; idx < 16 * DK_; idx += 256) {
        const int rr = idx >> 5;
        const int dd = idx & 31;
        out[(size_t)(b * S_ + rowBase + rr) * D_ + h * DK_ + dd] = ctxbuf[idx];
    }
}

// ---------------------------------------------------------------------------
extern "C" void kernel_launch(void* const* d_in, const int* in_sizes, int n_in,
                              void* d_out, int out_size, void* d_ws, size_t ws_size,
                              hipStream_t stream) {
    const float* x     = (const float*)d_in[0];
    const int*   xtype = (const int*)  d_in[1];
    const int*   mask  = (const int*)  d_in[2];
    const float* Wq    = (const float*)d_in[3];
    const float* bq    = (const float*)d_in[4];
    const float* Wk    = (const float*)d_in[5];
    const float* bk    = (const float*)d_in[6];
    const float* Wv    = (const float*)d_in[7];
    const float* bv    = (const float*)d_in[8];

    float* out   = (float*)d_out;                       // [B,S,D]
    float* pattn = out + (size_t)B_ * S_ * D_;          // [B,H,S,S]

    // workspace layout (f16):
    //   qws, kws : [b,h,s,dk]   vtws : [b,h,dk,s]     (3 x 4 MB)
    //   xmsk     : [T][NTOK, D] type-masked x copies  (16 MB)
    //   wqh/wkh/wvh : [T, D, D]                       (3 x 0.5 MB)
    _Float16* qws  = (_Float16*)d_ws;
    _Float16* kws  = qws + (size_t)NTOK * D_;
    _Float16* vtws = kws + (size_t)NTOK * D_;
    _Float16* xmsk = vtws + (size_t)NTOK * D_;
    _Float16* wqh  = xmsk + (size_t)T_ * NTOK * D_;
    _Float16* wkh  = wqh + (size_t)T_ * D_ * D_;
    _Float16* wvh  = wkh + (size_t)T_ * D_ * D_;

    const int xn4 = (NTOK * D_) / 4;            // 524288
    const int wn4 = (T_ * D_ * D_) / 4;         // 65536
    xmask_f16_kernel<<<dim3((xn4 + 255) / 256), 256, 0, stream>>>(x, xtype, xmsk, xn4);
    cvt_f16_kernel<<<dim3((wn4 + 255) / 256), 256, 0, stream>>>(Wq, wqh, wn4);
    cvt_f16_kernel<<<dim3((wn4 + 255) / 256), 256, 0, stream>>>(Wk, wkh, wn4);
    cvt_f16_kernel<<<dim3((wn4 + 255) / 256), 256, 0, stream>>>(Wv, wvh, wn4);

    hgs_qkv_kernel<<<dim3(D_ / 16, NTOK / 16 / 8, 1), 256, 0, stream>>>(
        xmsk, xtype, wqh, bq, wkh, bk, wvh, bv, qws, kws, vtws);

    const size_t smemBytes = (size_t)SMEM_FLOATS * sizeof(float);  // ~131 KB of 320 KB LDS
    (void)hipFuncSetAttribute((const void*)hgs_attn_kernel,
                              hipFuncAttributeMaxDynamicSharedMemorySize, (int)smemBytes);
    hgs_attn_kernel<<<dim3(S_ / 16, H_, B_), 256, smemBytes, stream>>>(
        qws, kws, vtws, mask, out, pattn);
}
